// MultiheadAttention_43147241455631
// MI455X (gfx1250) — matrix operand order repaired and checked by reading, not compiled, not run on hardware
//
#include <hip/hip_runtime.h>

// ---------------------------------------------------------------------------
// MI455X (gfx1250, wave32) fused multi-head-attention forward.
// All matmuls on v_wmma_f32_16x16x32_bf16; K / W / V^T tile staging via the
// Tensor Data Mover (tensor_load_to_lds + s_wait_tensorcnt).
// ---------------------------------------------------------------------------

typedef __attribute__((ext_vector_type(16))) __bf16     v16bf;
typedef __attribute__((ext_vector_type(8)))  __bf16     v8bf;
typedef __attribute__((ext_vector_type(8)))  float      v8f;
typedef __attribute__((ext_vector_type(4)))  float      v4f;
typedef __attribute__((ext_vector_type(4)))  unsigned   v4u;
typedef __attribute__((ext_vector_type(2)))  unsigned   v2u;
typedef __attribute__((ext_vector_type(8)))  int        v8i;
typedef __attribute__((ext_vector_type(4)))  int        v4i;

#define BATCH 8
#define SEQ   2048
#define HID   512
#define MROWS (BATCH * SEQ)

// fp32 -> bf16, round-to-nearest-even
__device__ __forceinline__ unsigned short f2bf(float f) {
  unsigned u = __builtin_bit_cast(unsigned, f);
  u += 0x7FFFu + ((u >> 16) & 1u);
  return (unsigned short)(u >> 16);
}
__device__ __forceinline__ unsigned pack2(float a, float b) {
  return (unsigned)f2bf(a) | ((unsigned)f2bf(b) << 16);
}
// Build a 16-element bf16 fragment from two 16-byte pieces (A/B operand
// layouts on CDNA5 split each lane's 16 values into two 8-value chunks).
__device__ __forceinline__ v16bf frag16(const unsigned short* p0,
                                        const unsigned short* p1) {
  v8bf lo = __builtin_bit_cast(v8bf, *(const v4u*)p0);
  v8bf hi = __builtin_bit_cast(v8bf, *(const v4u*)p1);
  return __builtin_shufflevector(lo, hi, 0, 1, 2, 3, 4, 5, 6, 7,
                                         8, 9, 10, 11, 12, 13, 14, 15);
}

// ---------------------------------------------------------------------------
// TDM: async 2D bf16 tile load global->LDS with LDS row padding.
// D# per CDNA5 ISA ch.8: group0 {count, lds_addr, global_addr, type=2},
// group1 {data_size=2B, pad cfg, tensor dims, tile dims, row stride}.
// Issue from one wave; completion via TENSORcnt.
// ---------------------------------------------------------------------------
__device__ __forceinline__ void tdm_load_2d_bf16(
    unsigned lds_byte_off, const void* gptr, unsigned tensor_w,
    unsigned tensor_h, unsigned long long row_stride_elems, unsigned tile_w,
    unsigned tile_h, unsigned pad_interval_code, unsigned pad_amount_code) {
  unsigned long long ga = (unsigned long long)(uintptr_t)gptr;
  v4u g0 = {
      1u,                                   // count=1 (user descriptor)
      lds_byte_off,                         // lds_addr [63:32]
      (unsigned)(ga & 0xFFFFFFFFu),         // global_addr [95:64]
      (unsigned)((ga >> 32) & 0x01FFFFFFu)  // global_addr [120:96]
          | (2u << 30)                      // type = 2 ("image")
  };
  v8i g1 = {
      (int)((1u << 16)                       // data_size: 2 bytes
            | (1u << 20)                     // pad_enable
            | (pad_interval_code << 22) | (pad_amount_code << 25)),
      (int)((tensor_w & 0xFFFFu) << 16),     // tensor_dim0 lo16 (barrier=0)
      (int)(((tensor_w >> 16) & 0xFFFFu) | ((tensor_h & 0xFFFFu) << 16)),
      (int)(((tensor_h >> 16) & 0xFFFFu) | ((tile_w & 0xFFFFu) << 16)),
      (int)(tile_h & 0xFFFFu),               // tile_dim1 (tile_dim2 = 0)
      (int)(row_stride_elems & 0xFFFFFFFFull),
      (int)((row_stride_elems >> 32) & 0xFFFFull),
      0};
  v4i z4 = {0, 0, 0, 0};
#if __clang_major__ >= 23
  v8i z8 = {0, 0, 0, 0, 0, 0, 0, 0};
  __builtin_amdgcn_tensor_load_to_lds(g0, g1, z4, z4, z8, 0);
#else
  __builtin_amdgcn_tensor_load_to_lds(g0, g1, z4, z4, 0);
#endif
}

// ---------------------------------------------------------------------------
// Prep: fp32 weight -> bf16 (RNE).  262144 elems, 4 per thread.
// ---------------------------------------------------------------------------
__global__ void __launch_bounds__(256)
cvt_w_kernel(const float* __restrict__ W, unsigned short* __restrict__ Wb) {
  int i = (blockIdx.x * 256 + threadIdx.x) * 4;
  v4f v = *(const v4f*)(W + i);
  v2u o = {pack2(v[0], v[1]), pack2(v[2], v[3])};
  *(v2u*)(Wb + i) = o;
}

// ---------------------------------------------------------------------------
// Prep: mask[row] = sign(|sum_d X[row][d]|)  (one wave32 per row)
// ---------------------------------------------------------------------------
__global__ void __launch_bounds__(256)
rowmask_kernel(const float* __restrict__ X, float* __restrict__ mask) {
  int row  = blockIdx.x * 8 + (threadIdx.x >> 5);
  int lane = threadIdx.x & 31;
  const float* src = X + (size_t)row * HID;
  float s = 0.f;
  for (int i = lane; i < HID; i += 32) s += src[i];
#pragma unroll
  for (int m = 16; m >= 1; m >>= 1) s += __shfl_xor(s, m, 32);
  if (lane == 0) mask[row] = (s != 0.f) ? 1.f : 0.f;
}

// ---------------------------------------------------------------------------
// V transpose (bf16): Vt[b][h][s] = Vb[b][s][h].  64x64 LDS tiles, both
// global sides fully coalesced (b128).  One-time 32 MB of traffic.
// ---------------------------------------------------------------------------
__global__ void __launch_bounds__(256)
vtrans_kernel(const unsigned short* __restrict__ Vb,
              unsigned short* __restrict__ Vt) {
  __shared__ __align__(16) unsigned short T[64][72];
  const int bb = blockIdx.z;
  const int sbase = blockIdx.x * 64;
  const int hbase = blockIdx.y * 64;
  const unsigned short* src = Vb + (size_t)bb * SEQ * HID;
  unsigned short* dst = Vt + (size_t)bb * HID * SEQ;
  const int tid = threadIdx.x;
#pragma unroll
  for (int it = 0; it < 2; ++it) {
    int idx = it * 256 + tid;
    int row = idx >> 3, c0 = (idx & 7) * 8;  // row = s local, c0 = h local
    *(v4u*)&T[row][c0] =
        *(const v4u*)(src + (size_t)(sbase + row) * HID + hbase + c0);
  }
  __syncthreads();
#pragma unroll
  for (int it = 0; it < 2; ++it) {
    int idx = it * 256 + tid;
    int orow = idx >> 3, c0 = (idx & 7) * 8;  // orow = h local, c0 = s local
    v4u o = {0u, 0u, 0u, 0u};
#pragma unroll
    for (int e = 0; e < 4; ++e)
      o[e] = (unsigned)T[c0 + 2 * e][orow] |
             ((unsigned)T[c0 + 2 * e + 1][orow] << 16);
    *(v4u*)(dst + (size_t)(hbase + orow) * SEQ + sbase + c0) = o;
  }
}

// ---------------------------------------------------------------------------
// Projection GEMM: Out[m][n] = bf16( sum_k X[m][k] * W[n][k] + bias[n] )
// Tile 128x128x32, 128 threads = 4 waves, 32 rows per wave (2 A fragments):
// each B fragment feeds two WMMAs, amortizing LDS-load waits.
// A staged manually (fp32->bf16 on the fly); B (weights, bf16) staged by the
// TDM: rows of 32 elems = 16 DWORDs (interval code 3), pad 4 DWORDs (code 3).
// ---------------------------------------------------------------------------
__global__ void __launch_bounds__(128)
proj_kernel(const float* __restrict__ X, const unsigned short* __restrict__ Wb,
            const float* __restrict__ bias, unsigned short* __restrict__ Out) {
  __shared__ __align__(16) unsigned short As[128][40];  // [m][k] bf16 (+pad)
  __shared__ __align__(16) unsigned short Bs[128][40];  // [n][k] bf16 (+pad)

  const int tid  = threadIdx.x;
  const int wave = tid >> 5, lane = tid & 31;
  const int half = lane >> 4, l16 = lane & 15;
  const int mbase = blockIdx.x * 128;
  const int nbase = blockIdx.y * 128;
  const unsigned bs_off = (unsigned)(uintptr_t)(const void*)&Bs[0][0];

  v8f acc[2][8];
#pragma unroll
  for (int g = 0; g < 2; ++g)
#pragma unroll
    for (int t = 0; t < 8; ++t)
      acc[g][t] = (v8f){0.f, 0.f, 0.f, 0.f, 0.f, 0.f, 0.f, 0.f};

  for (int k0 = 0; k0 < HID; k0 += 32) {
    __syncthreads();
    {  // stage A: one 32-float row per thread, fp32 -> bf16
      const float* src = X + (size_t)(mbase + tid) * HID + k0;
      v4f f0 = *(const v4f*)(src + 0);
      v4f f1 = *(const v4f*)(src + 4);
      v4f f2 = *(const v4f*)(src + 8);
      v4f f3 = *(const v4f*)(src + 12);
      v4f f4 = *(const v4f*)(src + 16);
      v4f f5 = *(const v4f*)(src + 20);
      v4f f6 = *(const v4f*)(src + 24);
      v4f f7 = *(const v4f*)(src + 28);
      v4u q0 = {pack2(f0[0], f0[1]), pack2(f0[2], f0[3]),
                pack2(f1[0], f1[1]), pack2(f1[2], f1[3])};
      v4u q1 = {pack2(f2[0], f2[1]), pack2(f2[2], f2[3]),
                pack2(f3[0], f3[1]), pack2(f3[2], f3[3])};
      v4u q2 = {pack2(f4[0], f4[1]), pack2(f4[2], f4[3]),
                pack2(f5[0], f5[1]), pack2(f5[2], f5[3])};
      v4u q3 = {pack2(f6[0], f6[1]), pack2(f6[2], f6[3]),
                pack2(f7[0], f7[1]), pack2(f7[2], f7[3])};
      *(v4u*)&As[tid][0]  = q0;
      *(v4u*)&As[tid][8]  = q1;
      *(v4u*)&As[tid][16] = q2;
      *(v4u*)&As[tid][24] = q3;
    }
    if (wave == 0) {  // stage B via TDM: 128 rows x 32 bf16, stride 512
      tdm_load_2d_bf16(bs_off, Wb + (size_t)nbase * HID + k0,
                       /*tensor_w=*/HID, /*tensor_h=*/128,
                       /*row_stride=*/512ull, /*tile_w=*/32, /*tile_h=*/128,
                       /*pad_interval(16DW)=*/3, /*pad_amount(4DW)=*/3);
      __builtin_amdgcn_s_wait_tensorcnt(0);
    }
    __syncthreads();

    // A fragments: lane<16 rows M=l16 K{0..15}; lane>=16 K{16..31}
    const unsigned short* ap0 = &As[wave * 32 + l16][half * 16];
    const unsigned short* ap1 = &As[wave * 32 + 16 + l16][half * 16];
    v16bf af0 = frag16(ap0, ap0 + 8);
    v16bf af1 = frag16(ap1, ap1 + 8);
    v16bf bfr[8];
#pragma unroll
    for (int t = 0; t < 8; ++t) {  // preload all B fragments
      const unsigned short* bp = &Bs[t * 16 + l16][half * 16];
      bfr[t] = frag16(bp, bp + 8);
    }
#pragma unroll
    for (int t = 0; t < 8; ++t) {  // each B fragment -> 2 WMMAs
      acc[0][t] = __builtin_amdgcn_wmma_f32_16x16x32_bf16(
          false, af0, false, bfr[t], (short)0, acc[0][t], false, false);
      acc[1][t] = __builtin_amdgcn_wmma_f32_16x16x32_bf16(
          false, af1, false, bfr[t], (short)0, acc[1][t], false, false);
    }
  }

  // epilogue: + bias, -> bf16.  C layout: row = r + 8*half, col = l16.
#pragma unroll
  for (int g = 0; g < 2; ++g)
#pragma unroll
    for (int t = 0; t < 8; ++t) {
      int col = nbase + t * 16 + l16;
      float bv = bias[col];
#pragma unroll
      for (int r = 0; r < 8; ++r) {
        int row = mbase + wave * 32 + g * 16 + r + half * 8;
        Out[(size_t)row * HID + col] = f2bf(acc[g][t][r] + bv);
      }
    }
}

// ---------------------------------------------------------------------------
// Flash attention: per block, 64 q-rows (4 waves x 16 rows), kv tiles of 128.
//   S = Q K^T / sqrt(H)  (+causal & key masks) -> online softmax -> O += P V
//   out = O/l * qmask + queries
// K chunks (64-wide) and V^T chunks (128-wide) staged by the TDM with LDS
// row padding to the 136-ushort stride.
// ---------------------------------------------------------------------------
#define FKV  128
#define LSTR 136  // row stride (ushorts): 272 B, 16-B aligned, bank-skewed

__global__ void __launch_bounds__(128)
flash_kernel(const unsigned short* __restrict__ Qb,
             const unsigned short* __restrict__ Kb,
             const unsigned short* __restrict__ Vt,
             const float* __restrict__ kmask, const float* __restrict__ qmask,
             const float* __restrict__ queries, float* __restrict__ out) {
  __shared__ __align__(16) unsigned short smem[192 * LSTR];  // 52 KB
  unsigned short* KV = smem;               // Ks [128][64] / Vs^T [128][128]
  unsigned short* Ps = smem + 128 * LSTR;  // per-wave P tiles [4*16][128]
  const unsigned kv_off = (unsigned)(uintptr_t)(const void*)KV;

  const int tid  = threadIdx.x;
  const int wave = tid >> 5, lane = tid & 31;
  const int half = lane >> 4, l16 = lane & 15;
  const int b     = blockIdx.x >> 5;
  const int qt    = blockIdx.x & 31;
  const int qbase = qt * 64;
  const size_t boff = (size_t)b * SEQ * HID;

  const int qrow0 = qbase + wave * 16;  // wave's first q row
  // per-lane Q row pointer for A fragments (bf16, straight from L2)
  const unsigned short* qrowp = Qb + boff + (size_t)(qrow0 + l16) * HID;

  float m_r[8], l_r[8];
#pragma unroll
  for (int r = 0; r < 8; ++r) { m_r[r] = -3.0e38f; l_r[r] = 0.f; }
  v8f o_acc[32];
#pragma unroll
  for (int t = 0; t < 32; ++t)
    o_acc[t] = (v8f){0.f, 0.f, 0.f, 0.f, 0.f, 0.f, 0.f, 0.f};

  const float sm_scale = 0.044194173824159216f;  // 1/sqrt(512)
  const float NEGC     = -4294967295.0f;
  const int jmax = (qbase + 63) >> 7;  // causal: last needed kv tile

  for (int j = 0; j <= jmax; ++j) {
    const int kvbase = j * FKV;
    v8f s_acc[8];
#pragma unroll
    for (int t = 0; t < 8; ++t)
      s_acc[t] = (v8f){0.f, 0.f, 0.f, 0.f, 0.f, 0.f, 0.f, 0.f};

    // ---- scores: S += Q[:,hc] K[:,hc]^T over 8 h-chunks of 64 ----
    for (int hc = 0; hc < 8; ++hc) {
      if (wave == 0) {  // TDM: K chunk 128 rows x 64 bf16, stride 512
        tdm_load_2d_bf16(kv_off,
                         Kb + boff + (size_t)kvbase * HID + (hc << 6),
                         /*tensor_w=*/HID, /*tensor_h=*/FKV,
                         /*row_stride=*/512ull, /*tile_w=*/64, /*tile_h=*/FKV,
                         /*pad_interval(32DW)=*/4, /*pad_amount(4DW)=*/3);
        __builtin_amdgcn_s_wait_tensorcnt(0);
      }
      __syncthreads();
#pragma unroll
      for (int kk = 0; kk < 2; ++kk) {
        int h = (hc << 6) + (kk << 5);
        v16bf afrag = frag16(qrowp + h + half * 16, qrowp + h + half * 16 + 8);
        v16bf bfr[8];
#pragma unroll
        for (int t = 0; t < 8; ++t) {
          const unsigned short* bp =
              &KV[(t * 16 + l16) * LSTR + (kk << 5) + half * 16];
          bfr[t] = frag16(bp, bp + 8);
        }
#pragma unroll
        for (int t = 0; t < 8; ++t)
          s_acc[t] = __builtin_amdgcn_wmma_f32_16x16x32_bf16(
              false, afrag, false, bfr[t], (short)0, s_acc[t], false, false);
      }
      __syncthreads();
    }

    // ---- masks + online softmax (C layout: row = r + 8*half, col = l16) ----
    float kmv[8];
#pragma unroll
    for (int t = 0; t < 8; ++t)
      kmv[t] = kmask[b * SEQ + kvbase + t * 16 + l16];

    float rowmax[8];
#pragma unroll
    for (int r = 0; r < 8; ++r) rowmax[r] = -3.0e38f;
#pragma unroll
    for (int t = 0; t < 8; ++t) {
      int col = kvbase + t * 16 + l16;
      bool kmz = (kmv[t] == 0.f);
#pragma unroll
      for (int r = 0; r < 8; ++r) {
        int qrow = qrow0 + r + half * 8;
        float s = s_acc[t][r] * sm_scale;
        s = (col > qrow || kmz) ? NEGC : s;
        s_acc[t][r] = s;
        rowmax[r] = fmaxf(rowmax[r], s);
      }
    }
#pragma unroll
    for (int r = 0; r < 8; ++r) {  // reduce within the 16-lane half-group
      float v = rowmax[r];
#pragma unroll
      for (int m = 8; m >= 1; m >>= 1) v = fmaxf(v, __shfl_xor(v, m, 32));
      rowmax[r] = v;
    }
    float scale_r[8];
#pragma unroll
    for (int r = 0; r < 8; ++r) {
      float mnew = fmaxf(m_r[r], rowmax[r]);
      scale_r[r] = __expf(m_r[r] - mnew);
      m_r[r] = mnew;
    }
    float rowsum[8];
#pragma unroll
    for (int r = 0; r < 8; ++r) rowsum[r] = 0.f;
#pragma unroll
    for (int t = 0; t < 8; ++t)
#pragma unroll
      for (int r = 0; r < 8; ++r) {
        float p = __expf(s_acc[t][r] - m_r[r]);
        s_acc[t][r] = p;
        rowsum[r] += p;
      }
#pragma unroll
    for (int r = 0; r < 8; ++r) {
      float v = rowsum[r];
#pragma unroll
      for (int m = 8; m >= 1; m >>= 1) v += __shfl_xor(v, m, 32);
      l_r[r] = l_r[r] * scale_r[r] + v;
    }
#pragma unroll
    for (int t = 0; t < 32; ++t)  // rescale O accumulators
#pragma unroll
      for (int r = 0; r < 8; ++r) o_acc[t][r] *= scale_r[r];

    // ---- P -> LDS (C layout -> A layout re-staging) ----
#pragma unroll
    for (int t = 0; t < 8; ++t)
#pragma unroll
      for (int r = 0; r < 8; ++r)
        Ps[(wave * 16 + r + half * 8) * LSTR + t * 16 + l16] =
            f2bf(s_acc[t][r]);
    __syncthreads();

    // ---- O += P V  over 4 h-chunks of 128 (V^T staged by TDM) ----
#pragma unroll
    for (int hc2 = 0; hc2 < 4; ++hc2) {
      if (wave == 0) {  // TDM: V^T chunk 128 h-rows x 128 kv, stride SEQ
        tdm_load_2d_bf16(
            kv_off,
            Vt + (size_t)b * HID * SEQ + (size_t)(hc2 * 128) * SEQ + kvbase,
            /*tensor_w=*/SEQ, /*tensor_h=*/128,
            /*row_stride=*/(unsigned long long)SEQ,
            /*tile_w=*/128, /*tile_h=*/128,
            /*pad_interval(64DW)=*/5, /*pad_amount(4DW)=*/3);
        __builtin_amdgcn_s_wait_tensorcnt(0);
      }
      __syncthreads();
#pragma unroll
      for (int kk = 0; kk < 4; ++kk) {
        const unsigned short* ap =
            &Ps[(wave * 16 + l16) * LSTR + (kk << 5) + half * 16];
        v16bf afrag = frag16(ap, ap + 8);
        v16bf bfr[8];
#pragma unroll
        for (int t = 0; t < 8; ++t) {
          const unsigned short* bp =
              &KV[(t * 16 + l16) * LSTR + (kk << 5) + half * 16];
          bfr[t] = frag16(bp, bp + 8);
        }
#pragma unroll
        for (int t = 0; t < 8; ++t)
          o_acc[hc2 * 8 + t] = __builtin_amdgcn_wmma_f32_16x16x32_bf16(
              false, afrag, false, bfr[t], (short)0, o_acc[hc2 * 8 + t],
              false, false);
      }
      __syncthreads();
    }
  }

  // ---- epilogue: out = O/l * qmask + queries ----
#pragma unroll
  for (int r = 0; r < 8; ++r) {
    int qrow = qrow0 + r + half * 8;
    float qm = qmask[b * SEQ + qrow];
    float inv_l = qm / l_r[r];
#pragma unroll
    for (int t = 0; t < 32; ++t) {
      size_t idx = boff + (size_t)qrow * HID + t * 16 + l16;
      out[idx] = o_acc[t][r] * inv_l + queries[idx];
    }
  }
}

// ---------------------------------------------------------------------------
extern "C" void kernel_launch(void* const* d_in, const int* in_sizes, int n_in,
                              void* d_out, int out_size, void* d_ws,
                              size_t ws_size, hipStream_t stream) {
  (void)in_sizes; (void)n_in; (void)out_size; (void)ws_size;
  const float* queries = (const float*)d_in[0];
  const float* keys    = (const float*)d_in[1];
  const float* Wq = (const float*)d_in[2];
  const float* bq = (const float*)d_in[3];
  const float* Wk = (const float*)d_in[4];
  const float* bk = (const float*)d_in[5];
  const float* Wv = (const float*)d_in[6];
  const float* bv = (const float*)d_in[7];
  float* out = (float*)d_out;

  // workspace carve (~68 MB): bf16 Q/K/V + V^T, bf16 weights, row masks
  const size_t qkv = (size_t)MROWS * HID;
  unsigned short* Qb  = (unsigned short*)d_ws;
  unsigned short* Kb  = Qb + qkv;
  unsigned short* Vb  = Kb + qkv;
  unsigned short* Wqb = Vb + qkv;
  unsigned short* Wkb = Wqb + (size_t)HID * HID;
  unsigned short* Wvb = Wkb + (size_t)HID * HID;
  float* kmaskp = (float*)(Wvb + (size_t)HID * HID);
  float* qmaskp = kmaskp + MROWS;
  unsigned short* Vtb = (unsigned short*)(qmaskp + MROWS);  // [B][H][S]

  cvt_w_kernel<<<256, 256, 0, stream>>>(Wq, Wqb);
  cvt_w_kernel<<<256, 256, 0, stream>>>(Wk, Wkb);
  cvt_w_kernel<<<256, 256, 0, stream>>>(Wv, Wvb);
  rowmask_kernel<<<2048, 256, 0, stream>>>(queries, qmaskp);
  rowmask_kernel<<<2048, 256, 0, stream>>>(keys, kmaskp);

  dim3 pgrid(MROWS / 128, HID / 128);
  proj_kernel<<<pgrid, 128, 0, stream>>>(queries, Wqb, bq, Qb);
  proj_kernel<<<pgrid, 128, 0, stream>>>(keys, Wkb, bk, Kb);
  proj_kernel<<<pgrid, 128, 0, stream>>>(keys, Wvb, bv, Vb);

  vtrans_kernel<<<dim3(SEQ / 64, HID / 64, BATCH), 256, 0, stream>>>(Vb, Vtb);

  flash_kernel<<<BATCH * (SEQ / 64), 128, 0, stream>>>(
      Qb, Kb, Vtb, kmaskp, qmaskp, queries, out);
}
